// MultiheadAttention_23948737643103
// MI455X (gfx1250) — matrix operand order repaired and checked by reading, not compiled, not run on hardware
//
#include <hip/hip_runtime.h>
#include <hip/hip_bf16.h>
#include <cstdint>

typedef _Float16 v16h __attribute__((ext_vector_type(16)));
typedef _Float16 v8h  __attribute__((ext_vector_type(8)));
typedef _Float16 v4h  __attribute__((ext_vector_type(4)));
typedef _Float16 v2h  __attribute__((ext_vector_type(2)));
typedef float    v8f  __attribute__((ext_vector_type(8)));

#define BS     16384
#define NCTX   10
#define WIDTH  768
#define NHEADS 12
#define HDIM   64
#define BB     8            // batches per workgroup
#define ROWS   (BB * NCTX)  // 80 rows = 5 WMMA M-tiles
#define LDX    776          // LDS stride (halves) for 768-wide tiles
#define LDQ    400          // LDS stride for per-head-pair qkv staging (384 used)
#define NW     8            // waves per block (wave32)

__device__ __forceinline__ v16h concat8(v8h a, v8h b) {
  return __builtin_shufflevector(a, b, 0, 1, 2, 3, 4, 5, 6, 7, 8, 9, 10, 11, 12, 13, 14, 15);
}

// A operand (16x32 f16, row-major source, leading dim ld in halves).
// lane<16: row m0+sub, K = {k..k+7, k+16..k+23}; lane>=16: same row, K + 8.
__device__ __forceinline__ v16h load_A(const _Float16* base, int ld, int m0, int k, int sub, int hi) {
  const _Float16* p = base + (m0 + sub) * ld + k + hi * 8;
  return concat8(*(const v8h*)p, *(const v8h*)(p + 16));
}

// B operand (32x16 f16) from row-major B^T (N x K, leading dim ld).
// lane<16: col n0+sub, K = {k..k+7, k+16..k+23}; lane>=16: same col, K + 8.
__device__ __forceinline__ v16h load_B(const _Float16* bt, int ld, int n0, int k, int sub, int hi) {
  const _Float16* p = bt + (size_t)(n0 + sub) * ld + k + hi * 8;
  return concat8(*(const v8h*)p, *(const v8h*)(p + 16));
}

__device__ __forceinline__ v8f wmma_f16(v16h a, v16h b, v8f c) {
  return __builtin_amdgcn_wmma_f32_16x16x32_f16(false, a, false, b, (short)0, c, false, false);
}

// Opaque zero (SGPR): defeats LICM/CSE of LDS reads across outer loops without
// disturbing addrspace inference or scheduling inside the loop.
__device__ __forceinline__ int opaque_zero() {
  int z = 0;
  asm volatile("" : "+s"(z));
  return z;
}

// ---------------------------------------------------------------------------
// Kernel 0: transpose weights once into workspace (L2-resident thereafter)
// ---------------------------------------------------------------------------
__global__ __launch_bounds__(256) void transpose_w(const _Float16* __restrict__ wqkv,
                                                   const _Float16* __restrict__ wproj,
                                                   _Float16* __restrict__ wtq,
                                                   _Float16* __restrict__ wtp) {
  int idx = blockIdx.x * 256 + threadIdx.x;
  if (idx < 3 * WIDTH * WIDTH) {  // 2304 x 768
    int n = idx / WIDTH, k = idx % WIDTH;
    wtq[idx] = wqkv[(size_t)k * (3 * WIDTH) + n];
  }
  if (idx < WIDTH * WIDTH) {      // 768 x 768
    int n = idx / WIDTH, k = idx % WIDTH;
    wtp[idx] = wproj[(size_t)k * WIDTH + n];
  }
}

// ---------------------------------------------------------------------------
// Kernel 1: fused QKV GEMM (WMMA) + per-head attention; qkv never hits HBM.
// ---------------------------------------------------------------------------
__global__ __launch_bounds__(256) void qkv_attn_kernel(const _Float16* __restrict__ x,
                                                       const uint8_t* __restrict__ kpm,
                                                       const _Float16* __restrict__ aw,
                                                       const _Float16* __restrict__ bqkv,
                                                       const _Float16* __restrict__ wtq,
                                                       _Float16* __restrict__ attn_out) {
  __shared__ _Float16 lds_x[ROWS * LDX];   // 124,160 B : X tile, A operand source
  __shared__ _Float16 qkv_s[ROWS * LDQ];   //  64,000 B : q|k|v for 2 heads
  __shared__ float scrS[NW][NCTX][12];     // raw scores per wave
  __shared__ float scrW[NW][NCTX][12];     // final weights per wave

  const int tid  = threadIdx.x;
  const int wave = tid >> 5;
  const int lane = tid & 31;
  const int sub  = lane & 15;
  const int hi   = lane >> 4;
  const int b0   = blockIdx.x * BB;

  // ---- stage X tile (80 x 768 fp16) into LDS, b128 coalesced
  {
    const _Float16* xsrc = x + (size_t)b0 * NCTX * WIDTH;
    for (int i = tid; i < ROWS * (WIDTH / 8); i += 256) {
      int row = i / (WIDTH / 8);
      int col = (i % (WIDTH / 8)) * 8;
      *(v8h*)&lds_x[row * LDX + col] = *(const v8h*)&xsrc[(size_t)row * WIDTH + col];
    }
  }
  __syncthreads();

  for (int hp = 0; hp < NHEADS / 2; ++hp) {
    // ---- QKV GEMM for head pair hp: qkv_s[80][384] = Xtile @ Wqkv[:, hp*384 .. +384) + b
    {
      const int kop = opaque_zero();           // block LICM across hp iterations
      const int nbase = wave * 3;              // this wave's 3 N-tiles (of 24)
      v8f acc[3][5] = {};
#pragma unroll 1
      for (int k = 0; k < WIDTH; k += 32) {
        v16h Av[5];
#pragma unroll
        for (int m = 0; m < 5; ++m) Av[m] = load_A(lds_x, LDX, m * 16, k + kop, sub, hi);
#pragma unroll
        for (int nt = 0; nt < 3; ++nt) {
          v16h Bv = load_B(wtq, WIDTH, hp * 384 + (nbase + nt) * 16, k, sub, hi);
#pragma unroll
          for (int m = 0; m < 5; ++m) acc[nt][m] = wmma_f16(Av[m], Bv, acc[nt][m]);
        }
      }
      // C layout: VGPR r -> row m*16 + r + 8*hi, col = nl + sub
#pragma unroll
      for (int nt = 0; nt < 3; ++nt) {
        const int nl = (nbase + nt) * 16;
        const float bias = (float)bqkv[hp * 384 + nl + sub];
#pragma unroll
        for (int m = 0; m < 5; ++m)
#pragma unroll
          for (int r = 0; r < 8; ++r)
            qkv_s[(m * 16 + r + hi * 8) * LDQ + nl + sub] = (_Float16)(acc[nt][m][r] + bias);
      }
    }
    __syncthreads();

    // ---- attention: wave handles batch bb = wave, both heads of the pair
    const int bb = wave;
    const int rb = bb * NCTX;
    for (int hh = 0; hh < 2; ++hh) {
      const int head = hp * 2 + hh;
      const int cb   = hh * 192;  // q at cb, k at cb+64, v at cb+128

      // raw scores S[t][s] = (q.k) * scale^2, scale^2 = 1/sqrt(64) = 0.125
      for (int e = lane; e < 100; e += 32) {
        int t = e / 10, s = e % 10;
        const _Float16* qp = &qkv_s[(rb + t) * LDQ + cb];
        const _Float16* kp = &qkv_s[(rb + s) * LDQ + cb + 64];
        float d = 0.f;
#pragma unroll
        for (int c = 0; c < HDIM; c += 4) {
          v4h qv = *(const v4h*)(qp + c);
          v4h kv = *(const v4h*)(kp + c);
          d += (float)qv[0] * (float)kv[0] + (float)qv[1] * (float)kv[1] +
               (float)qv[2] * (float)kv[2] + (float)qv[3] * (float)kv[3];
        }
        d *= 0.125f;
        if (kpm[(b0 + bb) * NCTX + s]) d = -__builtin_inff();
        scrS[wave][t][s] = d;
      }
      __syncthreads();

      // softmax(f32) -> fp16, * (1 + 0.2*a), renormalize in fp16 semantics
      if (lane < NCTX) {
        const int t = lane;
        float r[NCTX];
        float mx = -__builtin_inff();
#pragma unroll
        for (int s = 0; s < NCTX; ++s) { r[s] = scrS[wave][t][s]; mx = fmaxf(mx, r[s]); }
        float sum = 0.f;
#pragma unroll
        for (int s = 0; s < NCTX; ++s) { r[s] = __expf(r[s] - mx); sum += r[s]; }
        // faithful torch repeat: a[b,h] = attention_weights[(b*12 + h) % BS]
        const int ab = ((b0 + bb) * NHEADS + head) % BS;
        const _Float16* ap = aw + (size_t)ab * (NCTX * NCTX) + t * NCTX;
        float wv[NCTX];
        float wsum = 0.f;
#pragma unroll
        for (int s = 0; s < NCTX; ++s) {
          _Float16 p16 = (_Float16)(r[s] / sum);
          _Float16 f16 = (_Float16)(1.0f + 0.2f * (float)ap[s]);
          _Float16 w16 = (_Float16)((float)p16 * (float)f16);
          wv[s] = (float)w16;
          wsum += wv[s];
        }
        const float d16 = (float)(_Float16)wsum;
#pragma unroll
        for (int s = 0; s < NCTX; ++s)
          scrW[wave][t][s] = (float)(_Float16)(wv[s] / d16);
      }
      __syncthreads();

      // O[t][c] = sum_s W[t][s] * V[s][c]; each lane owns 2 channels
      {
        const int c = lane * 2;
        const _Float16* vb = &qkv_s[rb * LDQ + cb + 128 + c];
        _Float16* ob = attn_out + (size_t)(b0 + bb) * NCTX * WIDTH + head * HDIM + c;
        for (int t = 0; t < NCTX; ++t) {
          float a0 = 0.f, a1 = 0.f;
#pragma unroll
          for (int s = 0; s < NCTX; ++s) {
            float w = scrW[wave][t][s];
            v2h vv = *(const v2h*)(vb + s * LDQ);
            a0 += w * (float)vv[0];
            a1 += w * (float)vv[1];
          }
          v2h o;
          o[0] = (_Float16)a0;
          o[1] = (_Float16)a1;
          *(v2h*)(ob + (size_t)t * WIDTH) = o;
        }
      }
      __syncthreads();  // protect qkv_s before next head-pair GEMM overwrites it
    }
  }
}

// ---------------------------------------------------------------------------
// Kernel 2: output projection GEMM (WMMA) + bias -> d_out (fp16)
// ---------------------------------------------------------------------------
__global__ __launch_bounds__(256) void proj_kernel(const _Float16* __restrict__ ain,
                                                   const _Float16* __restrict__ wtp,
                                                   const _Float16* __restrict__ bproj,
                                                   _Float16* __restrict__ out) {
  __shared__ _Float16 lds_a[ROWS * LDX];
  const int tid  = threadIdx.x;
  const int wave = tid >> 5;
  const int lane = tid & 31;
  const int sub  = lane & 15;
  const int hi   = lane >> 4;
  const size_t r0 = (size_t)blockIdx.x * ROWS;

  const _Float16* src = ain + r0 * WIDTH;
  for (int i = tid; i < ROWS * (WIDTH / 8); i += 256) {
    int row = i / (WIDTH / 8);
    int col = (i % (WIDTH / 8)) * 8;
    *(v8h*)&lds_a[row * LDX + col] = *(const v8h*)&src[(size_t)row * WIDTH + col];
  }
  __syncthreads();

  // 48 N-tiles / 8 waves = 6 per wave, processed as 2 groups of 3 (15 accs each)
  for (int g = 0; g < 2; ++g) {
    const int kop = opaque_zero();   // block LICM across the two groups
    const int nbase = wave * 6 + g * 3;
    v8f acc[3][5] = {};
#pragma unroll 1
    for (int k = 0; k < WIDTH; k += 32) {
      v16h Av[5];
#pragma unroll
      for (int m = 0; m < 5; ++m) Av[m] = load_A(lds_a, LDX, m * 16, k + kop, sub, hi);
#pragma unroll
      for (int nt = 0; nt < 3; ++nt) {
        v16h Bv = load_B(wtp, WIDTH, (nbase + nt) * 16, k, sub, hi);
#pragma unroll
        for (int m = 0; m < 5; ++m) acc[nt][m] = wmma_f16(Av[m], Bv, acc[nt][m]);
      }
    }
#pragma unroll
    for (int nt = 0; nt < 3; ++nt) {
      const int n0 = (nbase + nt) * 16;
      const float bias = (float)bproj[n0 + sub];
#pragma unroll
      for (int m = 0; m < 5; ++m)
#pragma unroll
        for (int r = 0; r < 8; ++r)
          out[(r0 + m * 16 + r + hi * 8) * WIDTH + n0 + sub] = (_Float16)(acc[nt][m][r] + bias);
    }
  }
}

extern "C" void kernel_launch(void* const* d_in, const int* in_sizes, int n_in,
                              void* d_out, int out_size, void* d_ws, size_t ws_size,
                              hipStream_t stream) {
  (void)in_sizes; (void)n_in; (void)out_size; (void)ws_size;
  const _Float16* x     = (const _Float16*)d_in[0];
  const uint8_t*  kpm   = (const uint8_t*)d_in[1];
  const _Float16* aw    = (const _Float16*)d_in[2];
  const _Float16* wqkv  = (const _Float16*)d_in[3];
  const _Float16* bqkv  = (const _Float16*)d_in[4];
  const _Float16* wproj = (const _Float16*)d_in[5];
  const _Float16* bproj = (const _Float16*)d_in[6];
  _Float16* out = (_Float16*)d_out;

  char* ws = (char*)d_ws;
  _Float16* wtq  = (_Float16*)ws;                                       // 2304*768 fp16
  _Float16* wtp  = (_Float16*)(ws + (size_t)3 * WIDTH * WIDTH * 2);     //  768*768 fp16
  _Float16* attn = (_Float16*)(ws + (size_t)3 * WIDTH * WIDTH * 2 +
                               (size_t)WIDTH * WIDTH * 2);              // 163840*768 fp16

  transpose_w<<<(3 * WIDTH * WIDTH + 255) / 256, 256, 0, stream>>>(wqkv, wproj, wtq, wtp);
  qkv_attn_kernel<<<BS / BB, 256, 0, stream>>>(x, kpm, aw, bqkv, wtq, attn);
  proj_kernel<<<(BS * NCTX) / ROWS, 256, 0, stream>>>(attn, wtp, bproj, out);
}
